// NATLayer_81724637708877
// MI455X (gfx1250) — compile-verified, matrix operand order checked by reading, not run on hardware
//
#include <hip/hip_runtime.h>
#include <hip/hip_bf16.h>
#include <math.h>

typedef __attribute__((ext_vector_type(16))) _Float16 v16h;
typedef __attribute__((ext_vector_type(8)))  _Float16 h8v;
typedef __attribute__((ext_vector_type(8)))  float    v8f;

#define Bd 2
#define Cd 96
#define Hd 128
#define Wd 128
#define HEADSd 4
#define DHd 24
#define HWd (Hd*Wd)
#define Nrow (Bd*HWd)
#define EPSf 1e-5f

__device__ __forceinline__ float geluf(float v) {
  return 0.5f * v * (1.f + erff(v * 0.70710678118654752f));
}
__device__ __forceinline__ float sigmf(float v) {
  return 1.f / (1.f + __expf(-v));
}

// fp32 -> fp16 weight pre-conversion (weights stay L2-resident in f16)
__global__ void k_cvt16(const float* __restrict__ in, _Float16* __restrict__ out, int n) {
  int t = blockIdx.x * blockDim.x + threadIdx.x;
  if (t < n) out[t] = (_Float16)in[t];
}

// ---------------------------------------------------------------------------
// WMMA GEMM, all-f16 operands: Y[row,o] = act( sum_k X[row,k]*Wt[o,k] + bias[o] )
// X: [Rows, Kdim] f16 row-major, Wt: [Odim, Kdim] f16 row-major.
// Each wave computes a 16(M) x 96(N) strip = 6 N-tiles. Per 32-wide K step:
// 2 b128 loads for the A fragment, 12 clause-able b128 loads preloading all
// 6 B fragments, then 6 back-to-back v_wmma_f32_16x16x32_f16.
// Requires (Odim/16) % 6 == 0 (true for Odim in {96, 288, 384}).
// ACT: 0=none 1=relu 2=gelu; OUT16: 1 -> f16 output, 0 -> f32 output.
// ---------------------------------------------------------------------------
template <int ACT, int OUT16>
__global__ void k_gemm_wmma(const _Float16* __restrict__ X,
                            const _Float16* __restrict__ Wt,
                            const float* __restrict__ bias,
                            void* __restrict__ Yv,
                            int Kdim, int Odim, int strips) {
  constexpr int NT = 6;
  int wid = blockIdx.x * (blockDim.x >> 5) + (threadIdx.x >> 5);
  if (wid >= strips) return;
  int lane = threadIdx.x & 31;
  int stripsN = (Odim >> 4) / NT;
  int tM = wid / stripsN;
  int tN = (wid - tM * stripsN) * NT;
  int r16  = lane & 15;
  int half = lane >> 4;

  const _Float16* xrow  = X  + (size_t)(tM * 16 + r16) * Kdim;
  const _Float16* wbase = Wt + (size_t)(tN * 16 + r16) * Kdim;
  const size_t wstep = (size_t)16 * Kdim;

  v8f acc[NT];
#pragma unroll
  for (int n = 0; n < NT; ++n) acc[n] = (v8f){};

  for (int k0 = 0; k0 < Kdim; k0 += 32) {
    if (k0 + 32 < Kdim) {
      __builtin_prefetch(xrow + k0 + 32, 0, 0);   // gfx1250 global_prefetch
    }
    // A fragment: lane covers k = k0 + half*8 + 0..7 and k0 + 16 + half*8 + 0..7
    h8v alo = *(const h8v*)(xrow + k0 + half * 8);
    h8v ahi = *(const h8v*)(xrow + k0 + 16 + half * 8);
    // Preload all 6 B fragments (distinct registers -> loads pipeline)
    h8v bl[NT], bh[NT];
#pragma unroll
    for (int n = 0; n < NT; ++n) {
      const _Float16* wr = wbase + (size_t)n * wstep + k0;
      bl[n] = *(const h8v*)(wr + half * 8);
      bh[n] = *(const h8v*)(wr + 16 + half * 8);
    }
    v16h av;
#pragma unroll
    for (int j = 0; j < 8; ++j) { av[j] = alo[j]; av[8 + j] = ahi[j]; }
#pragma unroll
    for (int n = 0; n < NT; ++n) {
      v16h bv;
#pragma unroll
      for (int j = 0; j < 8; ++j) { bv[j] = bl[n][j]; bv[8 + j] = bh[n][j]; }
      acc[n] = __builtin_amdgcn_wmma_f32_16x16x32_f16(
          false, av, false, bv, (short)0, acc[n], false, false);
    }
  }

#pragma unroll
  for (int n = 0; n < NT; ++n) {
    int col = (tN + n) * 16 + r16;
    float bb = bias[col];
#pragma unroll
    for (int r = 0; r < 8; ++r) {
      int row = tM * 16 + r + (half << 3);
      float v = acc[n][r] + bb;
      if (ACT == 1) v = fmaxf(v, 0.f);
      else if (ACT == 2) v = geluf(v);
      if (OUT16) ((_Float16*)Yv)[(size_t)row * Odim + col] = (_Float16)v;
      else       ((float*)Yv)[(size_t)row * Odim + col] = v;
    }
  }
}

// ---------------------------------------------------------------------------
// RLN: per-sample mean/std over (C,H,W)
// ---------------------------------------------------------------------------
__global__ void k_rln_stats(const float* __restrict__ x, float* __restrict__ stats) {
  int b = blockIdx.x;
  const float* xb = x + (size_t)b * Cd * HWd;
  float s = 0.f, s2 = 0.f;
  for (int i = threadIdx.x; i < Cd * HWd; i += blockDim.x) {
    float v = xb[i]; s += v; s2 += v * v;
  }
  __shared__ float sh[256], sh2[256];
  sh[threadIdx.x] = s; sh2[threadIdx.x] = s2;
  __syncthreads();
  for (int o = 128; o > 0; o >>= 1) {
    if ((int)threadIdx.x < o) { sh[threadIdx.x] += sh[threadIdx.x + o]; sh2[threadIdx.x] += sh2[threadIdx.x + o]; }
    __syncthreads();
  }
  if (threadIdx.x == 0) {
    const float n = (float)(Cd * HWd);
    float mean = sh[0] / n;
    float var  = fmaxf(sh2[0] / n - mean * mean, 0.f);
    stats[b * 2]     = mean;
    stats[b * 2 + 1] = sqrtf(var + EPSf);
  }
}

__global__ void k_meta(const float* __restrict__ stats,
                       const float* __restrict__ m1w, const float* __restrict__ m1b,
                       const float* __restrict__ m2w, const float* __restrict__ m2b,
                       float* __restrict__ rs, float* __restrict__ rbv) {
  int t = blockIdx.x * blockDim.x + threadIdx.x;
  if (t >= Bd * Cd) return;
  int b = t / Cd, c = t - b * Cd;
  rs[t]  = stats[b * 2 + 1] * m1w[c] + m1b[c];
  rbv[t] = stats[b * 2]     * m2w[c] + m2b[c];
}

// NCHW f32 -> normalized NHWC f16 (feeds WMMA GEMMs directly)
__global__ void k_rln_apply(const float* __restrict__ xin, const float* __restrict__ stats,
                            const float* __restrict__ rw, const float* __restrict__ rb,
                            _Float16* __restrict__ xn) {
  int t = blockIdx.x * blockDim.x + threadIdx.x;
  if (t >= Bd * Cd * HWd) return;
  int c = t % Cd;
  int p = (t / Cd) % HWd;
  int b = t / (Cd * HWd);
  float mean = stats[b * 2], istd = 1.f / stats[b * 2 + 1];
  float v = xin[((size_t)b * Cd + c) * HWd + p];
  xn[t] = (_Float16)((v - mean) * istd * rw[c] + rb[c]);
}

// x_out(NCHW) = sc(NCHW) + y(NHWC f32)*rs[b,c] + rb[b,c]
__global__ void k_combine(const float* __restrict__ sc, const float* __restrict__ ynhwc,
                          const float* __restrict__ rs, const float* __restrict__ rbv,
                          float* __restrict__ xout) {
  int t = blockIdx.x * blockDim.x + threadIdx.x;
  if (t >= Bd * Cd * HWd) return;
  int p = t % HWd;
  int c = (t / HWd) % Cd;
  int b = t / (HWd * Cd);
  xout[t] = sc[t] + ynhwc[((size_t)b * HWd + p) * Cd + c] * rs[b * Cd + c] + rbv[b * Cd + c];
}

// load 24 f16 (16B-aligned) as 3 x b128 into fp32 registers
__device__ __forceinline__ void load24(const _Float16* __restrict__ p, float* d) {
  h8v a0 = *(const h8v*)p;
  h8v a1 = *(const h8v*)(p + 8);
  h8v a2 = *(const h8v*)(p + 16);
#pragma unroll
  for (int j = 0; j < 8; ++j) { d[j] = (float)a0[j]; d[8 + j] = (float)a1[j]; d[16 + j] = (float)a2[j]; }
}

// ---------------------------------------------------------------------------
// NAT neighborhood attention; one thread per (b, head, y, x).
// qkv: [Nrow, 3*Cd] f16 rows (q | k | v, each heads*dh); out: f16 [Nrow, Cd]
// ---------------------------------------------------------------------------
template <int KS, int DS>
__global__ void k_nat(const _Float16* __restrict__ qkv, const float* __restrict__ rpb,
                      _Float16* __restrict__ out) {
  int gid = blockIdx.x * blockDim.x + threadIdx.x;
  if (gid >= Bd * HEADSd * HWd) return;
  int xi   = gid % Wd;
  int yi   = (gid / Wd) % Hd;
  int head = (gid / HWd) % HEADSd;
  int b    = gid / (HWd * HEADSd);

  const int R = 2 * KS - 1;
  int row = (b * Hd + yi) * Wd + xi;

  float q[DHd];
  load24(qkv + (size_t)row * (3 * Cd) + head * DHd, q);
  const float scale = 0.2041241452319315f;  // 24^-0.5
#pragma unroll
  for (int j = 0; j < DHd; ++j) q[j] *= scale;

  // clamped dilated window indexing (NATTEN semantics)
  int gy = yi % DS, iiy = yi / DS;
  int Lgy = (Hd - gy + DS - 1) / DS;
  int sy = min(max(iiy - KS / 2, 0), Lgy - KS);
  int gx = xi % DS, iix = xi / DS;
  int Lgx = (Wd - gx + DS - 1) / DS;
  int sx = min(max(iix - KS / 2, 0), Lgx - KS);

  float attn[KS * KS];
  float mx = -3.4e38f;
#pragma unroll
  for (int u = 0; u < KS; ++u) {
    int ky = gy + DS * (sy + u);
    int ry = sy + u - iiy + (KS - 1);
#pragma unroll
    for (int v = 0; v < KS; ++v) {
      int kx = gx + DS * (sx + v);
      int rx = sx + v - iix + (KS - 1);
      float kv[DHd];
      load24(qkv + (size_t)((b * Hd + ky) * Wd + kx) * (3 * Cd) + Cd + head * DHd, kv);
      float s = rpb[(head * R + ry) * R + rx];
#pragma unroll
      for (int j = 0; j < DHd; ++j) s += q[j] * kv[j];
      attn[u * KS + v] = s;
      mx = fmaxf(mx, s);
    }
  }
  float sum = 0.f;
#pragma unroll
  for (int i = 0; i < KS * KS; ++i) { attn[i] = __expf(attn[i] - mx); sum += attn[i]; }
  float inv = 1.f / sum;

  float o[DHd];
#pragma unroll
  for (int j = 0; j < DHd; ++j) o[j] = 0.f;
#pragma unroll
  for (int u = 0; u < KS; ++u) {
    int ky = gy + DS * (sy + u);
#pragma unroll
    for (int v = 0; v < KS; ++v) {
      int kx = gx + DS * (sx + v);
      float vv[DHd];
      load24(qkv + (size_t)((b * Hd + ky) * Wd + kx) * (3 * Cd) + 2 * Cd + head * DHd, vv);
      float a = attn[u * KS + v] * inv;
#pragma unroll
      for (int j = 0; j < DHd; ++j) o[j] += a * vv[j];
    }
  }
  _Float16* op = out + (size_t)row * Cd + head * DHd;
  h8v o0, o1, o2;
#pragma unroll
  for (int j = 0; j < 8; ++j) {
    o0[j] = (_Float16)o[j]; o1[j] = (_Float16)o[8 + j]; o2[j] = (_Float16)o[16 + j];
  }
  *(h8v*)op = o0; *(h8v*)(op + 8) = o1; *(h8v*)(op + 16) = o2;
}

// ---------------------------------------------------------------------------
// BatchNorm stats (per channel over B,H,W) and apply
// ---------------------------------------------------------------------------
__global__ void k_bn_stats(const float* __restrict__ x, float* __restrict__ mu,
                           float* __restrict__ var) {
  int c = blockIdx.x;
  float s = 0.f, s2 = 0.f;
  for (int i = threadIdx.x; i < Bd * HWd; i += blockDim.x) {
    int b = i / HWd, p = i - b * HWd;
    float v = x[((size_t)b * Cd + c) * HWd + p];
    s += v; s2 += v * v;
  }
  __shared__ float sh[256], sh2[256];
  sh[threadIdx.x] = s; sh2[threadIdx.x] = s2;
  __syncthreads();
  for (int o = 128; o > 0; o >>= 1) {
    if ((int)threadIdx.x < o) { sh[threadIdx.x] += sh[threadIdx.x + o]; sh2[threadIdx.x] += sh2[threadIdx.x + o]; }
    __syncthreads();
  }
  if (threadIdx.x == 0) {
    const float n = (float)(Bd * HWd);
    float m = sh[0] / n;
    mu[c]  = m;
    var[c] = fmaxf(sh2[0] / n - m * m, 0.f);
  }
}

__global__ void k_bn_apply(const float* __restrict__ x, const float* __restrict__ mu,
                           const float* __restrict__ var, const float* __restrict__ g,
                           const float* __restrict__ bb, float* __restrict__ y) {
  int t = blockIdx.x * blockDim.x + threadIdx.x;
  if (t >= Bd * Cd * HWd) return;
  int c = (t / HWd) % Cd;
  y[t] = (x[t] - mu[c]) * rsqrtf(var[c] + EPSf) * g[c] + bb[c];
}

// pixel attention: pa[b,p] = sigmoid(w2 . gelu(W1 x + b1) + b2)
__global__ void k_pa(const float* __restrict__ xbn, const float* __restrict__ w1,
                     const float* __restrict__ b1, const float* __restrict__ w2,
                     const float* __restrict__ b2, float* __restrict__ pa) {
  int pid = blockIdx.x * blockDim.x + threadIdx.x;
  if (pid >= Bd * HWd) return;
  int b = pid / HWd, p = pid - b * HWd;
  const float* xp = xbn + (size_t)b * Cd * HWd + p;
  float xv[Cd];
#pragma unroll 8
  for (int c = 0; c < Cd; ++c) xv[c] = xp[(size_t)c * HWd];
  float acc = b2[0];
  for (int o = 0; o < Cd / 8; ++o) {
    float a = b1[o];
    const float* wr = w1 + o * Cd;
#pragma unroll 8
    for (int c = 0; c < Cd; ++c) a += wr[c] * xv[c];
    acc += w2[o] * geluf(a);
  }
  pa[pid] = sigmf(acc);
}

__global__ void k_pa_mul(const float* __restrict__ xbn, const float* __restrict__ pa,
                         float* __restrict__ y) {
  int t = blockIdx.x * blockDim.x + threadIdx.x;
  if (t >= Bd * Cd * HWd) return;
  int p = t % HWd;
  int b = t / (HWd * Cd);
  y[t] = xbn[t] * pa[b * HWd + p];
}

// global average pool per (b,c)
__global__ void k_ca_pool(const float* __restrict__ x, float* __restrict__ ga) {
  int bc = blockIdx.x;
  const float* xp = x + (size_t)bc * HWd;
  float s = 0.f;
  for (int i = threadIdx.x; i < HWd; i += blockDim.x) s += xp[i];
  __shared__ float sh[256];
  sh[threadIdx.x] = s;
  __syncthreads();
  for (int o = 128; o > 0; o >>= 1) {
    if ((int)threadIdx.x < o) sh[threadIdx.x] += sh[threadIdx.x + o];
    __syncthreads();
  }
  if (threadIdx.x == 0) ga[bc] = sh[0] / (float)HWd;
}

__global__ void k_ca1(const float* __restrict__ ga, const float* __restrict__ w1,
                      const float* __restrict__ b1, float* __restrict__ h) {
  int t = blockIdx.x * blockDim.x + threadIdx.x;
  if (t >= Bd * Cd) return;
  int b = t / Cd, o = t - b * Cd;
  float a = b1[o];
  for (int c = 0; c < Cd; ++c) a += w1[o * Cd + c] * ga[b * Cd + c];
  h[t] = geluf(a);
}

__global__ void k_ca2(const float* __restrict__ h, const float* __restrict__ w2,
                      const float* __restrict__ b2, float* __restrict__ ca) {
  int t = blockIdx.x * blockDim.x + threadIdx.x;
  if (t >= Bd * Cd) return;
  int b = t / Cd, o = t - b * Cd;
  float a = b2[o];
  for (int c = 0; c < Cd; ++c) a += w2[o * Cd + c] * h[b * Cd + c];
  ca[t] = sigmf(a);
}

// NCHW f32 scaled by ca[b,c] -> NHWC f16 rows for the mlp2 GEMMs
__global__ void k_scale_nhwc(const float* __restrict__ x, const float* __restrict__ ca,
                             _Float16* __restrict__ y) {
  int t = blockIdx.x * blockDim.x + threadIdx.x;
  if (t >= Bd * Cd * HWd) return;
  int p = t % HWd;
  int c = (t / HWd) % Cd;
  int b = t / (HWd * Cd);
  y[((size_t)b * HWd + p) * Cd + c] = (_Float16)(x[t] * ca[b * Cd + c]);
}

// out(NCHW) = sc2(NCHW, in d_out already) + m2(NHWC f32)
__global__ void k_final(const float* __restrict__ m2, float* __restrict__ out) {
  int t = blockIdx.x * blockDim.x + threadIdx.x;
  if (t >= Bd * Cd * HWd) return;
  int p = t % HWd;
  int c = (t / HWd) % Cd;
  int b = t / (HWd * Cd);
  out[t] = out[t] + m2[((size_t)b * HWd + p) * Cd + c];
}

// ---------------------------------------------------------------------------
extern "C" void kernel_launch(void* const* d_in, const int* in_sizes, int n_in,
                              void* d_out, int out_size, void* d_ws, size_t ws_size,
                              hipStream_t stream) {
  (void)in_sizes; (void)n_in; (void)out_size; (void)ws_size;
  const float* x0     = (const float*)d_in[0];
  const float* rln_w  = (const float*)d_in[1];
  const float* rln_b  = (const float*)d_in[2];
  const float* m1w    = (const float*)d_in[3];
  const float* m1b    = (const float*)d_in[4];
  const float* m2w    = (const float*)d_in[5];
  const float* m2b    = (const float*)d_in[6];
  const float* qkv1_w = (const float*)d_in[7];
  const float* qkv1_b = (const float*)d_in[8];
  const float* rpb1   = (const float*)d_in[9];
  const float* proj1_w= (const float*)d_in[10];
  const float* proj1_b= (const float*)d_in[11];
  const float* qkv2_w = (const float*)d_in[12];
  const float* qkv2_b = (const float*)d_in[13];
  const float* rpb2   = (const float*)d_in[14];
  const float* proj2_w= (const float*)d_in[15];
  const float* proj2_b= (const float*)d_in[16];
  const float* mlp_w1 = (const float*)d_in[17];
  const float* mlp_b1 = (const float*)d_in[18];
  const float* mlp_w2 = (const float*)d_in[19];
  const float* mlp_b2 = (const float*)d_in[20];
  const float* bn_g   = (const float*)d_in[21];
  const float* bn_b   = (const float*)d_in[22];
  const float* pa_w1  = (const float*)d_in[23];
  const float* pa_b1  = (const float*)d_in[24];
  const float* pa_w2  = (const float*)d_in[25];
  const float* pa_b2  = (const float*)d_in[26];
  const float* ca_w1  = (const float*)d_in[27];
  const float* ca_b1  = (const float*)d_in[28];
  const float* ca_w2  = (const float*)d_in[29];
  const float* ca_b2  = (const float*)d_in[30];
  const float* m2_w1  = (const float*)d_in[31];
  const float* m2_b1  = (const float*)d_in[32];
  const float* m2_w2  = (const float*)d_in[33];
  const float* m2_b2  = (const float*)d_in[34];

  const size_t NC  = (size_t)Nrow * Cd;
  const size_t N4C = (size_t)Nrow * 4 * Cd;
  float* w0 = (float*)d_ws;     // f16 xn / f32 xbn / f32 m2-out      [NC f32]
  float* w1 = w0 + NC;          // f16 qkv / f16 mlp-hidden / f16 m2h [N4C f32]
  float* w2 = w1 + N4C;         // f16 nat-out / f32 x-nat2 / f32 xpa [NC f32]
  float* w3 = w2 + NC;          // f32 proj/mlp outs / f16 m2-in      [NC f32]
  float* w4 = w3 + NC;          // x after block1 (sc for blk2)       [NC f32]
  float* sm = w4 + NC;
  float* stats = sm;            // 4
  float* rs    = sm + 8;
  float* rbv   = rs + Bd * Cd;
  float* bnmu  = rbv + Bd * Cd;
  float* bnvar = bnmu + Cd;
  float* pabuf = bnvar + Cd;    // 32768
  float* ga    = pabuf + Bd * HWd;
  float* cah   = ga + Bd * Cd;
  float* cav   = cah + Bd * Cd;
  // f16 weight cache (converted once per launch)
  _Float16* hq1 = (_Float16*)(cav + Bd * Cd);
  _Float16* hp1 = hq1 + 3 * Cd * Cd;
  _Float16* hq2 = hp1 + Cd * Cd;
  _Float16* hp2 = hq2 + 3 * Cd * Cd;
  _Float16* hm1 = hp2 + Cd * Cd;
  _Float16* hm2 = hm1 + 4 * Cd * Cd;
  _Float16* h21 = hm2 + 4 * Cd * Cd;
  _Float16* h22 = h21 + Cd * Cd;
  float* outp  = (float*)d_out;

  const int EW = 256;
  const int tot = Bd * Cd * HWd;
  dim3 eb(EW), eg((tot + EW - 1) / EW);
  dim3 mg((Bd * Cd + EW - 1) / EW);
  dim3 natg((Bd * HEADSd * HWd + EW - 1) / EW);
  auto cvt = [&](const float* src, _Float16* dst, int n) {
    k_cvt16<<<(n + EW - 1) / EW, EW, 0, stream>>>(src, dst, n);
  };
  auto strips = [](int rows, int od) { return (rows / 16) * ((od / 16) / 6); };

  // ---- weight f16 pre-conversion (tiny; stays in L2 for the whole run) ----
  cvt(qkv1_w, hq1, 3 * Cd * Cd);
  cvt(proj1_w, hp1, Cd * Cd);
  cvt(qkv2_w, hq2, 3 * Cd * Cd);
  cvt(proj2_w, hp2, Cd * Cd);
  cvt(mlp_w1, hm1, 4 * Cd * Cd);
  cvt(mlp_w2, hm2, 4 * Cd * Cd);
  cvt(m2_w1, h21, Cd * Cd);
  cvt(m2_w2, h22, Cd * Cd);

  _Float16* xnh  = (_Float16*)w0;
  _Float16* qkvh = (_Float16*)w1;
  _Float16* nath = (_Float16*)w2;
  _Float16* hidh = (_Float16*)w1;
  _Float16* m2in = (_Float16*)w3;
  _Float16* m2h  = (_Float16*)w1;

  // ================= Block 1: RLN + NAT(7,1) =================
  k_rln_stats<<<Bd, 256, 0, stream>>>(x0, stats);
  k_meta<<<mg, eb, 0, stream>>>(stats, m1w, m1b, m2w, m2b, rs, rbv);
  k_rln_apply<<<eg, eb, 0, stream>>>(x0, stats, rln_w, rln_b, xnh);
  { int t = strips(Nrow, 3 * Cd);
    k_gemm_wmma<0, 1><<<(t + 7) / 8, 256, 0, stream>>>(xnh, hq1, qkv1_b, qkvh, Cd, 3 * Cd, t); }
  k_nat<7, 1><<<natg, eb, 0, stream>>>(qkvh, rpb1, nath);
  { int t = strips(Nrow, Cd);
    k_gemm_wmma<0, 0><<<(t + 7) / 8, 256, 0, stream>>>(nath, hp1, proj1_b, w3, Cd, Cd, t); }
  k_combine<<<eg, eb, 0, stream>>>(x0, w3, rs, rbv, w4);

  // ================= Block 2: RLN + NAT(5,8), shared shortcut w4 =============
  k_rln_stats<<<Bd, 256, 0, stream>>>(w4, stats);
  k_meta<<<mg, eb, 0, stream>>>(stats, m1w, m1b, m2w, m2b, rs, rbv);
  k_rln_apply<<<eg, eb, 0, stream>>>(w4, stats, rln_w, rln_b, xnh);
  { int t = strips(Nrow, 3 * Cd);
    k_gemm_wmma<0, 1><<<(t + 7) / 8, 256, 0, stream>>>(xnh, hq2, qkv2_b, qkvh, Cd, 3 * Cd, t); }
  k_nat<5, 8><<<natg, eb, 0, stream>>>(qkvh, rpb2, nath);
  { int t = strips(Nrow, Cd);
    k_gemm_wmma<0, 0><<<(t + 7) / 8, 256, 0, stream>>>(nath, hp2, proj2_b, w3, Cd, Cd, t); }
  k_combine<<<eg, eb, 0, stream>>>(w4, w3, rs, rbv, w2);  // x after NAT2 -> w2 (f32)

  // MLP (residual uses sc = w4)
  k_rln_stats<<<Bd, 256, 0, stream>>>(w2, stats);
  k_meta<<<mg, eb, 0, stream>>>(stats, m1w, m1b, m2w, m2b, rs, rbv);
  k_rln_apply<<<eg, eb, 0, stream>>>(w2, stats, rln_w, rln_b, xnh);
  { int t = strips(Nrow, 4 * Cd);
    k_gemm_wmma<1, 1><<<(t + 7) / 8, 256, 0, stream>>>(xnh, hm1, mlp_b1, hidh, Cd, 4 * Cd, t); }
  { int t = strips(Nrow, Cd);
    k_gemm_wmma<0, 0><<<(t + 7) / 8, 256, 0, stream>>>(hidh, hm2, mlp_b2, w3, 4 * Cd, Cd, t); }
  k_combine<<<eg, eb, 0, stream>>>(w4, w3, rs, rbv, outp);  // x after MLP = sc2 -> d_out

  // ================= BN + PA + CA + MLP2 =================
  k_bn_stats<<<Cd, 256, 0, stream>>>(outp, bnmu, bnvar);
  k_bn_apply<<<eg, eb, 0, stream>>>(outp, bnmu, bnvar, bn_g, bn_b, w0);
  k_pa<<<(Bd * HWd + EW - 1) / EW, eb, 0, stream>>>(w0, pa_w1, pa_b1, pa_w2, pa_b2, pabuf);
  k_pa_mul<<<eg, eb, 0, stream>>>(w0, pabuf, w2);
  k_ca_pool<<<Bd * Cd, 256, 0, stream>>>(w2, ga);
  k_ca1<<<mg, eb, 0, stream>>>(ga, ca_w1, ca_b1, cah);
  k_ca2<<<mg, eb, 0, stream>>>(cah, ca_w2, ca_b2, cav);
  k_scale_nhwc<<<eg, eb, 0, stream>>>(w2, cav, m2in);
  { int t = strips(Nrow, Cd);
    k_gemm_wmma<2, 1><<<(t + 7) / 8, 256, 0, stream>>>(m2in, h21, m2_b1, m2h, Cd, Cd, t); }
  { int t = strips(Nrow, Cd);
    k_gemm_wmma<0, 0><<<(t + 7) / 8, 256, 0, stream>>>(m2h, h22, m2_b2, w0, Cd, Cd, t); }
  k_final<<<eg, eb, 0, stream>>>(w0, outp);
}
